// SparseMOE_21741124452487
// MI455X (gfx1250) — matrix operand order repaired and checked by reading, not compiled, not run on hardware
//
#include <hip/hip_runtime.h>

// ---------------- problem constants ----------------
#define T_TOK   8192          // B*S
#define D_EMB   1024
#define D_FF    4096
#define N_EXP   8
#define MAXROWS 17408         // T_TOK*2 + N_EXP*128 padding, multiple of 128

#define MT  128               // M tile
#define NT  128               // N tile
#define KS  64                // K step
#define LDT 72                // LDS row stride (elements), 144B = 9*16B

typedef unsigned short u16;
typedef unsigned int   u32;

typedef __attribute__((ext_vector_type(16))) __bf16 bf16x16;
typedef __attribute__((ext_vector_type(8)))  float  f32x8;

union Frag { uint4 u[2]; bf16x16 v; };

// ---- CDNA5 async global->LDS path (guarded: falls back to sync ds stores) ----
#if defined(__HIP_DEVICE_COMPILE__) && __has_builtin(__builtin_amdgcn_global_load_async_to_lds_b128)
#define HAVE_ASYNC 1
#else
#define HAVE_ASYNC 0
#endif

#if HAVE_ASYNC
// From the compiler diagnostic: param0 is AS(1) pointer to 'int __vector_size__(16)'.
typedef int v4i __attribute__((vector_size(16)));
typedef __attribute__((address_space(1))) v4i* glb_v4p;
typedef __attribute__((address_space(3))) v4i* lds_v4p;
#define TO_GLB(p) ((glb_v4p)(unsigned long long)(p))
#define TO_LDS(p) ((lds_v4p)(unsigned int)(unsigned long long)(p))
#endif

__device__ __forceinline__ void ld_tile(const u16* __restrict__ g, u16* __restrict__ l) {
#if HAVE_ASYNC
#pragma unroll
    for (int j = 0; j < 4; j++)
        __builtin_amdgcn_global_load_async_to_lds_b128(TO_GLB(g + 8 * j), TO_LDS(l + 8 * j), 0, 0);
#else
    const uint4* gp = (const uint4*)g;
    uint4*       lp = (uint4*)l;
    lp[0] = gp[0]; lp[1] = gp[1]; lp[2] = gp[2]; lp[3] = gp[3];
#endif
}

__device__ __forceinline__ void wait_async8() {
#if HAVE_ASYNC
#if __has_builtin(__builtin_amdgcn_s_wait_asynccnt)
    __builtin_amdgcn_s_wait_asynccnt(8);
#else
    asm volatile("s_wait_asynccnt 8" ::: "memory");
#endif
#endif
}
__device__ __forceinline__ void wait_async0() {
#if HAVE_ASYNC
#if __has_builtin(__builtin_amdgcn_s_wait_asynccnt)
    __builtin_amdgcn_s_wait_asynccnt(0);
#else
    asm volatile("s_wait_asynccnt 0" ::: "memory");
#endif
#endif
}

__device__ __forceinline__ u16 f2bf(float f) {
    u32 x = __float_as_uint(f);
    u32 r = x + 0x7fffu + ((x >> 16) & 1u);   // round-to-nearest-even
    return (u16)(r >> 16);
}

// ---------------- init: zero out, meta, entries, zero-row of x16 ----------------
__global__ void k_init(float* __restrict__ out, int* __restrict__ entries,
                       int* __restrict__ meta, u16* __restrict__ xzero) {
    int i = blockIdx.x * 256 + threadIdx.x;
    if (i < T_TOK * D_EMB) out[i] = 0.f;
    if (i < MAXROWS)       entries[i] = -1;
    if (i < 32)            meta[i] = 0;
    if (i < D_EMB)         xzero[i] = 0;
}

// ---------------- f32 -> bf16 (x) ----------------
__global__ void k_cvt_bf16(const float* __restrict__ src, u16* __restrict__ dst, int n4) {
    int i = blockIdx.x * 256 + threadIdx.x;
    if (i >= n4) return;
    float4 f = ((const float4*)src)[i];
    u32 lo = (u32)f2bf(f.x) | ((u32)f2bf(f.y) << 16);
    u32 hi = (u32)f2bf(f.z) | ((u32)f2bf(f.w) << 16);
    uint2 o; o.x = lo; o.y = hi;
    ((uint2*)dst)[i] = o;
}

// ---------------- transpose+convert: src[z][R][C] f32 -> dst[z][C][R] bf16 ----------
__global__ void k_transpose_bf16(const float* __restrict__ src, u16* __restrict__ dst,
                                 int R, int C) {
    __shared__ u16 tile[32][33];
    int z = blockIdx.z;
    const float* s = src + (size_t)z * R * C;
    u16*         d = dst + (size_t)z * R * C;
    int cb = blockIdx.x * 32, rb = blockIdx.y * 32;
    int tx = threadIdx.x, ty = threadIdx.y;
    for (int j = ty; j < 32; j += 8)
        tile[j][tx] = f2bf(s[(size_t)(rb + j) * C + cb + tx]);
    __syncthreads();
    for (int j = ty; j < 32; j += 8)
        d[(size_t)(cb + j) * R + rb + tx] = tile[tx][j];
}

// ---------------- router: one wave per token ----------------
__global__ void __launch_bounds__(256) k_router(
    const float* __restrict__ x, const float* __restrict__ noise_u,
    const float* __restrict__ wg, const float* __restrict__ bg,
    const float* __restrict__ wn, const float* __restrict__ bn,
    float* __restrict__ gSlot, int* __restrict__ topE, int* __restrict__ meta) {
    int lane = threadIdx.x & 31;
    int wid  = threadIdx.x >> 5;
    int t = blockIdx.x * 8 + wid;

    float ag[N_EXP], an[N_EXP];
#pragma unroll
    for (int e = 0; e < N_EXP; e++) { ag[e] = 0.f; an[e] = 0.f; }

    const float* xr = x + (size_t)t * D_EMB;
    for (int i = 0; i < D_EMB / 32; i++) {
        int dd = i * 32 + lane;
        float xv = xr[dd];
        const float* gr = wg + dd * N_EXP;
        const float* nr = wn + dd * N_EXP;
#pragma unroll
        for (int e = 0; e < N_EXP; e++) {
            ag[e] += xv * gr[e];
            an[e] += xv * nr[e];
        }
    }
#pragma unroll
    for (int off = 16; off >= 1; off >>= 1) {
#pragma unroll
        for (int e = 0; e < N_EXP; e++) {
            ag[e] += __shfl_xor(ag[e], off, 32);
            an[e] += __shfl_xor(an[e], off, 32);
        }
    }
    if (lane == 0) {
        const float* nu = noise_u + (size_t)t * N_EXP;
        float noisy[N_EXP];
#pragma unroll
        for (int e = 0; e < N_EXP; e++) {
            float lg = ag[e] + bg[e];
            float nl = an[e] + bn[e];
            float sp = (nl > 30.f) ? nl : log1pf(expf(nl));
            noisy[e] = lg + nu[e] * sp;
        }
        int e0 = 0;
#pragma unroll
        for (int e = 1; e < N_EXP; e++) if (noisy[e] > noisy[e0]) e0 = e;
        int e1 = (e0 == 0) ? 1 : 0;
#pragma unroll
        for (int e = 0; e < N_EXP; e++)
            if (e != e0 && noisy[e] > noisy[e1]) e1 = e;
        float ex = expf(noisy[e1] - noisy[e0]);
        float g0 = 1.f / (1.f + ex);
        float g1 = ex / (1.f + ex);
        topE[t * 2 + 0] = e0; gSlot[t * 2 + 0] = g0;
        topE[t * 2 + 1] = e1; gSlot[t * 2 + 1] = g1;
        atomicAdd(&meta[e0], 1);
        atomicAdd(&meta[e1], 1);
    }
}

// ---------------- prefix scan of padded expert counts ----------------
__global__ void k_scan(int* __restrict__ meta) {
    if (threadIdx.x == 0 && blockIdx.x == 0) {
        int off = 0;
        for (int e = 0; e < N_EXP; e++) {
            meta[16 + e] = off;
            int c = meta[e];
            off += ((c + MT - 1) / MT) * MT;
        }
        meta[24] = off;   // padded total rows
    }
}

// ---------------- scatter slots into expert segments ----------------
__global__ void k_scatter(const int* __restrict__ topE, int* __restrict__ meta,
                          int* __restrict__ entries) {
    int s = blockIdx.x * 256 + threadIdx.x;
    if (s >= T_TOK * 2) return;
    int e = topE[s];
    int pos = meta[16 + e] + atomicAdd(&meta[8 + e], 1);
    entries[pos] = s;
}

// ---------------- FC1: H = relu(Xg * W1_e + b1_e)  (bf16 out) ----------------
__global__ void __launch_bounds__(256) k_fc1(
    const u16* __restrict__ x16, const u16* __restrict__ w1t,
    const float* __restrict__ b1, const int* __restrict__ meta,
    const int* __restrict__ entries, u16* __restrict__ Hbuf) {
    __shared__ __align__(16) u16 As[2][MT * LDT];
    __shared__ __align__(16) u16 Bs[2][NT * LDT];
    __shared__ int rowSlot[MT];

    const int* offs = meta + 16;
    int total = offs[8];
    int row0 = blockIdx.y * MT;
    if (row0 >= total) return;
    int e = 0;
#pragma unroll
    for (int i = 1; i < N_EXP; i++) if (row0 >= offs[i]) e = i;

    int tid = threadIdx.x;
    if (tid < MT) rowSlot[tid] = entries[row0 + tid];
    __syncthreads();

    int lane = tid & 31, wid = tid >> 5;
    int wm = wid >> 1, wn = wid & 1;
    int n0 = blockIdx.x * NT;

    f32x8 zero = {0.f, 0.f, 0.f, 0.f, 0.f, 0.f, 0.f, 0.f};
    f32x8 acc[2][4];
#pragma unroll
    for (int sm = 0; sm < 2; sm++)
#pragma unroll
        for (int sn = 0; sn < 4; sn++) acc[sm][sn] = zero;

    int lr = tid >> 1;          // 0..127
    int lc = (tid & 1) * 32;    // 0 or 32
    int slot = rowSlot[lr];
    int tok  = (slot >= 0) ? (slot >> 1) : T_TOK;      // T_TOK = zero row
    const u16* aSrc = x16 + (size_t)tok * D_EMB + lc;
    const u16* bSrc = w1t + ((size_t)e * D_FF + (n0 + lr)) * D_EMB + lc;
    int ldsOff = lr * LDT + lc;

    ld_tile(aSrc, &As[0][ldsOff]);
    ld_tile(bSrc, &Bs[0][ldsOff]);

    const int STEPS = D_EMB / KS;
    for (int i = 0; i < STEPS; i++) {
        int cur = i & 1, nxt = cur ^ 1;
        __syncthreads();                       // everyone done reading buf[nxt]
        if (i + 1 < STEPS) {
            ld_tile(aSrc + (i + 1) * KS, &As[nxt][ldsOff]);
            ld_tile(bSrc + (i + 1) * KS, &Bs[nxt][ldsOff]);
            wait_async8();                     // current tile (first 8 in-order) done
        } else {
            wait_async0();
        }
        __syncthreads();                       // all waves have cur tile in LDS

#pragma unroll
        for (int kk = 0; kk < 2; kk++) {
            Frag fa[2], fb[4];
#pragma unroll
            for (int sm = 0; sm < 2; sm++) {
                int m = wm * 32 + sm * 16 + (lane & 15);
                const u16* p = &As[cur][m * LDT + kk * 32 + ((lane >> 4) << 4)];
                fa[sm].u[0] = *(const uint4*)p;
                fa[sm].u[1] = *(const uint4*)(p + 8);
            }
#pragma unroll
            for (int sn = 0; sn < 4; sn++) {
                int n = wn * 64 + sn * 16 + (lane & 15);
                const u16* p = &Bs[cur][n * LDT + kk * 32 + ((lane >> 4) << 4)];
                fb[sn].u[0] = *(const uint4*)p;
                fb[sn].u[1] = *(const uint4*)(p + 8);
            }
#pragma unroll
            for (int sm = 0; sm < 2; sm++)
#pragma unroll
                for (int sn = 0; sn < 4; sn++)
                    acc[sm][sn] = __builtin_amdgcn_wmma_f32_16x16x32_bf16(
                        false, fa[sm].v, false, fb[sn].v,
                        (short)0, acc[sm][sn], false, false);
        }
    }

#pragma unroll
    for (int sn = 0; sn < 4; sn++) {
        int nc = n0 + wn * 64 + sn * 16 + (lane & 15);
        float bias = b1[e * D_FF + nc];
#pragma unroll
        for (int sm = 0; sm < 2; sm++) {
#pragma unroll
            for (int v = 0; v < 8; v++) {
                int m = wm * 32 + sm * 16 + v + ((lane >> 4) << 3);
                float h = acc[sm][sn][v] + bias;
                h = h > 0.f ? h : 0.f;
                Hbuf[(size_t)(row0 + m) * D_FF + nc] = f2bf(h);
            }
        }
    }
}

// ---------------- FC2: y = H * W2_e + b2_e, gate-scale, atomic combine ----------
__global__ void __launch_bounds__(256) k_fc2(
    const u16* __restrict__ Hbuf, const u16* __restrict__ w2t,
    const float* __restrict__ b2, const float* __restrict__ gSlot,
    const int* __restrict__ meta, const int* __restrict__ entries,
    float* __restrict__ out) {
    __shared__ __align__(16) u16 As[2][MT * LDT];
    __shared__ __align__(16) u16 Bs[2][NT * LDT];
    __shared__ int rowSlot[MT];

    const int* offs = meta + 16;
    int total = offs[8];
    int row0 = blockIdx.y * MT;
    if (row0 >= total) return;
    int e = 0;
#pragma unroll
    for (int i = 1; i < N_EXP; i++) if (row0 >= offs[i]) e = i;

    int tid = threadIdx.x;
    if (tid < MT) rowSlot[tid] = entries[row0 + tid];
    __syncthreads();

    int lane = tid & 31, wid = tid >> 5;
    int wm = wid >> 1, wn = wid & 1;
    int n0 = blockIdx.x * NT;

    f32x8 zero = {0.f, 0.f, 0.f, 0.f, 0.f, 0.f, 0.f, 0.f};
    f32x8 acc[2][4];
#pragma unroll
    for (int sm = 0; sm < 2; sm++)
#pragma unroll
        for (int sn = 0; sn < 4; sn++) acc[sm][sn] = zero;

    int lr = tid >> 1;
    int lc = (tid & 1) * 32;
    const u16* aSrc = Hbuf + (size_t)(row0 + lr) * D_FF + lc;
    const u16* bSrc = w2t + ((size_t)e * D_EMB + (n0 + lr)) * D_FF + lc;
    int ldsOff = lr * LDT + lc;

    ld_tile(aSrc, &As[0][ldsOff]);
    ld_tile(bSrc, &Bs[0][ldsOff]);

    const int STEPS = D_FF / KS;
    for (int i = 0; i < STEPS; i++) {
        int cur = i & 1, nxt = cur ^ 1;
        __syncthreads();
        if (i + 1 < STEPS) {
            ld_tile(aSrc + (i + 1) * KS, &As[nxt][ldsOff]);
            ld_tile(bSrc + (i + 1) * KS, &Bs[nxt][ldsOff]);
            wait_async8();
        } else {
            wait_async0();
        }
        __syncthreads();

#pragma unroll
        for (int kk = 0; kk < 2; kk++) {
            Frag fa[2], fb[4];
#pragma unroll
            for (int sm = 0; sm < 2; sm++) {
                int m = wm * 32 + sm * 16 + (lane & 15);
                const u16* p = &As[cur][m * LDT + kk * 32 + ((lane >> 4) << 4)];
                fa[sm].u[0] = *(const uint4*)p;
                fa[sm].u[1] = *(const uint4*)(p + 8);
            }
#pragma unroll
            for (int sn = 0; sn < 4; sn++) {
                int n = wn * 64 + sn * 16 + (lane & 15);
                const u16* p = &Bs[cur][n * LDT + kk * 32 + ((lane >> 4) << 4)];
                fb[sn].u[0] = *(const uint4*)p;
                fb[sn].u[1] = *(const uint4*)(p + 8);
            }
#pragma unroll
            for (int sm = 0; sm < 2; sm++)
#pragma unroll
                for (int sn = 0; sn < 4; sn++)
                    acc[sm][sn] = __builtin_amdgcn_wmma_f32_16x16x32_bf16(
                        false, fa[sm].v, false, fb[sn].v,
                        (short)0, acc[sm][sn], false, false);
        }
    }

#pragma unroll
    for (int sn = 0; sn < 4; sn++) {
        int nc = n0 + wn * 64 + sn * 16 + (lane & 15);
        float bias = b2[e * D_EMB + nc];
#pragma unroll
        for (int sm = 0; sm < 2; sm++) {
#pragma unroll
            for (int v = 0; v < 8; v++) {
                int m = wm * 32 + sm * 16 + v + ((lane >> 4) << 3);
                int slot = rowSlot[m];
                if (slot >= 0) {
                    int tokIdx = slot >> 1;
                    float yv = (acc[sm][sn][v] + bias) * gSlot[slot];
                    atomicAdd(&out[(size_t)tokIdx * D_EMB + nc], yv);
                }
            }
        }
    }
}

// ---------------- host launcher ----------------
extern "C" void kernel_launch(void* const* d_in, const int* in_sizes, int n_in,
                              void* d_out, int out_size, void* d_ws, size_t ws_size,
                              hipStream_t stream) {
    (void)in_sizes; (void)n_in; (void)out_size; (void)ws_size;
    const float* x       = (const float*)d_in[0];
    const float* noise_u = (const float*)d_in[1];
    const float* wg      = (const float*)d_in[2];
    const float* bg      = (const float*)d_in[3];
    const float* wn      = (const float*)d_in[4];
    const float* bn      = (const float*)d_in[5];
    const float* w1      = (const float*)d_in[6];
    const float* b1      = (const float*)d_in[7];
    const float* w2      = (const float*)d_in[8];
    const float* b2      = (const float*)d_in[9];
    float* out = (float*)d_out;

    char* p = (char*)d_ws;
    auto take = [&](size_t bytes) {
        void* r = (void*)p;
        p += (bytes + 255) & ~(size_t)255;
        return r;
    };
    int*   topE    = (int*)take((size_t)T_TOK * 2 * sizeof(int));
    float* gSlot   = (float*)take((size_t)T_TOK * 2 * sizeof(float));
    int*   meta    = (int*)take(32 * sizeof(int));       // [0..7] counts, [8..15] cursors, [16..24] offsets
    int*   entries = (int*)take((size_t)MAXROWS * sizeof(int));
    u16*   x16     = (u16*)take((size_t)(T_TOK + 1) * D_EMB * 2);   // +1 zero row
    u16*   w1t     = (u16*)take((size_t)N_EXP * D_EMB * D_FF * 2);
    u16*   w2t     = (u16*)take((size_t)N_EXP * D_EMB * D_FF * 2);
    u16*   Hbuf    = (u16*)take((size_t)MAXROWS * D_FF * 2);

    k_init<<<(T_TOK * D_EMB) / 256, 256, 0, stream>>>(out, entries, meta,
                                                      x16 + (size_t)T_TOK * D_EMB);
    k_cvt_bf16<<<(T_TOK * D_EMB / 4 + 255) / 256, 256, 0, stream>>>(x, x16, T_TOK * D_EMB / 4);

    dim3 tb(32, 8);
    k_transpose_bf16<<<dim3(D_FF / 32, D_EMB / 32, N_EXP), tb, 0, stream>>>(w1, w1t, D_EMB, D_FF);
    k_transpose_bf16<<<dim3(D_EMB / 32, D_FF / 32, N_EXP), tb, 0, stream>>>(w2, w2t, D_FF, D_EMB);

    k_router<<<T_TOK / 8, 256, 0, stream>>>(x, noise_u, wg, bg, wn, bn, gSlot, topE, meta);
    k_scan<<<1, 32, 0, stream>>>(meta);
    k_scatter<<<(T_TOK * 2) / 256, 256, 0, stream>>>(topE, meta, entries);

    k_fc1<<<dim3(D_FF / NT, MAXROWS / MT), 256, 0, stream>>>(x16, w1t, b1, meta, entries, Hbuf);
    k_fc2<<<dim3(D_EMB / NT, MAXROWS / MT), 256, 0, stream>>>(Hbuf, w2t, b2, gSlot, meta, entries, out);
}
